// CMRHead_26001732010627
// MI455X (gfx1250) — matrix operand order repaired and checked by reading, not compiled, not run on hardware
//
#include <hip/hip_runtime.h>
#include <hip/hip_bf16.h>

// ---------------------------------------------------------------------------
// Problem constants (from reference): stage=6, bs=64, nq=300, C=256, OUT=229
// ---------------------------------------------------------------------------
#define NROWS   115200      // 6*64*300
#define CDIM    256
#define ODIM    229         // 24*9 + 10 + 3
#define OPAD    240         // 15 WMMA n-tiles of 16
#define BN_EPSF 1e-5f

typedef __attribute__((ext_vector_type(16))) _Float16 v16h;
typedef __attribute__((ext_vector_type(8)))  _Float16 v8h;
typedef __attribute__((ext_vector_type(4)))  _Float16 v4h;
typedef __attribute__((ext_vector_type(8)))  float    v8f;
typedef __attribute__((ext_vector_type(4)))  float    v4f;

static __device__ __forceinline__ v16h join8(v8h lo, v8h hi) {
    return __builtin_shufflevector(lo, hi, 0,1,2,3,4,5,6,7,8,9,10,11,12,13,14,15);
}

// ---------------------------------------------------------------------------
// Weight prep: W (f32, [C][ncols] row-major)  ->  Wt (f16, [npad][C], i.e.
// transposed N-major so a B fragment is 16 contiguous halves per lane half).
// Pad rows (n >= ncols) zero-filled.
// ---------------------------------------------------------------------------
__global__ void wprep_kernel(const float* __restrict__ W, _Float16* __restrict__ Wt,
                             int ncols, int npad) {
    int idx = blockIdx.x * blockDim.x + threadIdx.x;
    if (idx >= npad * CDIM) return;
    int n = idx >> 8;          // / 256
    int k = idx & 255;
    float v = (n < ncols) ? W[k * ncols + n] : 0.0f;
    Wt[(size_t)n * CDIM + k] = (_Float16)v;
}

// f32 -> f16 convert (x -> a0), 4 elements/thread
__global__ void cvt_kernel(const float* __restrict__ src, _Float16* __restrict__ dst, size_t n4) {
    size_t t = (size_t)blockIdx.x * blockDim.x + threadIdx.x;
    if (t >= n4) return;
    v4f x = *(const v4f*)(src + t * 4);
    v4h h; h.x = (_Float16)x.x; h.y = (_Float16)x.y; h.z = (_Float16)x.z; h.w = (_Float16)x.w;
    *(v4h*)(dst + t * 4) = h;
}

__global__ void zero_kernel(float* __restrict__ p, int n) {
    int t = blockIdx.x * blockDim.x + threadIdx.x;
    if (t < n) p[t] = 0.0f;
}

// ---------------------------------------------------------------------------
// GEMM: Z[M x ldz](f32) = A[M x 256](f16) @ Wt^T + bias, with optional
// per-column sum / sum-of-squares accumulation for batch-norm statistics.
// Wave = 16 rows x NT*16 cols; block = 4 waves = 64 rows; grid = NROWS/64.
// B (and A) fragments are explicitly double-buffered so the load for the next
// tile is in flight while the current WMMA executes (avoids s_wait_loadcnt 0
// in front of every v_wmma).
// ---------------------------------------------------------------------------
template<int NT, bool STATS>
__global__ __launch_bounds__(128)
void gemm_f16_kernel(const _Float16* __restrict__ A, const _Float16* __restrict__ Bw,
                     const float* __restrict__ bias, float* __restrict__ Z,
                     float* __restrict__ gsum, float* __restrict__ gsq,
                     int ldz, int ncols) {
    __shared__ float ssum[CDIM];
    __shared__ float ssq[CDIM];

    const int lane = threadIdx.x & 31;
    const int wave = threadIdx.x >> 5;
    const int hi   = lane >> 4;           // lane half (0/1)
    const int lm   = lane & 15;
    const int rowBase = blockIdx.x * 64 + wave * 16;
    const int arow = rowBase + lm;        // A-fragment row for this lane

    if constexpr (STATS) {
        for (int i = threadIdx.x; i < CDIM; i += blockDim.x) { ssum[i] = 0.0f; ssq[i] = 0.0f; }
        __syncthreads();
    }

    v8f acc[NT] = {};
    const _Float16* __restrict__ ap = A + (size_t)arow * CDIM;
    const _Float16* __restrict__ bbase = Bw + (size_t)lm * CDIM + hi * 16;

    // A fragment: 16 contiguous halves per lane half (k = 16*hi + s), matching B
    auto loadA = [&](int k) -> v16h {
        v8h a0 = *(const v8h*)(ap + k + hi * 16);
        v8h a1 = *(const v8h*)(ap + k + hi * 16 + 8);
        return join8(a0, a1);
    };
    // B fragment: 16 contiguous halves per lane half (k = 16*hi + s), matching A
    auto loadB = [&](int nt, int k) -> v16h {
        const _Float16* bp = bbase + (size_t)nt * (16 * CDIM) + k;
        v8h b0 = *(const v8h*)(bp);
        v8h b1 = *(const v8h*)(bp + 8);
        return join8(b0, b1);
    };

    v16h af   = loadA(0);
    v16h bcur = loadB(0, 0);
    for (int k = 0; k < CDIM; k += 32) {
        v16h afn = af;
        if (k + 32 < CDIM) afn = loadA(k + 32);          // prefetch next A frag
        #pragma unroll
        for (int nt = 0; nt < NT; ++nt) {
            v16h bnext = bcur;
            if (nt + 1 < NT)          bnext = loadB(nt + 1, k);      // prefetch next B
            else if (k + 32 < CDIM)   bnext = loadB(0, k + 32);
            acc[nt] = __builtin_amdgcn_wmma_f32_16x16x32_f16(
                false, af, false, bcur, (short)0, acc[nt], false, false);
            bcur = bnext;
        }
        af = afn;
    }

    // Epilogue: +bias, store f32, per-column stats via LDS atomics
    #pragma unroll
    for (int nt = 0; nt < NT; ++nt) {
        int col = nt * 16 + lm;
        float bv = (col < ncols) ? bias[col] : 0.0f;
        float s = 0.0f, q = 0.0f;
        #pragma unroll
        for (int v = 0; v < 8; ++v) {
            float zv = acc[nt][v] + bv;                 // C/D layout: M = v + 8*hi
            s += zv; q += zv * zv;
            if (col < ncols)
                Z[(size_t)(rowBase + v + hi * 8) * (size_t)ldz + col] = zv;
        }
        if constexpr (STATS) {
            atomicAdd(&ssum[col], s);
            atomicAdd(&ssq[col], q);
        }
    }

    if constexpr (STATS) {
        __syncthreads();
        for (int i = threadIdx.x; i < CDIM; i += blockDim.x) {
            atomicAdd(&gsum[i], ssum[i]);
            atomicAdd(&gsq[i],  ssq[i]);
        }
    }
}

// mean/var -> per-channel scale/shift
__global__ void bnfin_kernel(const float* __restrict__ sum, const float* __restrict__ sq,
                             const float* __restrict__ g, const float* __restrict__ be,
                             float* __restrict__ sc, float* __restrict__ sh, float invN) {
    int c = threadIdx.x;
    float m  = sum[c] * invN;
    float v  = sq[c] * invN - m * m;
    float rs = rsqrtf(v + BN_EPSF);
    float s  = g[c] * rs;
    sc[c] = s;
    sh[c] = be[c] - m * s;
}

// a = relu(z*sc + sh) -> f16, 4 elems/thread (column = flat index % 256)
__global__ void bnrelu_kernel(const float* __restrict__ Z,
                              const float* __restrict__ sc, const float* __restrict__ sh,
                              _Float16* __restrict__ Aout, size_t n4) {
    size_t t = (size_t)blockIdx.x * blockDim.x + threadIdx.x;
    if (t >= n4) return;
    size_t i4 = t * 4;
    int col = (int)(i4 & 255);
    v4f z = *(const v4f*)(Z + i4);
    v4h h;
    h.x = (_Float16)fmaxf(z.x * sc[col + 0] + sh[col + 0], 0.0f);
    h.y = (_Float16)fmaxf(z.y * sc[col + 1] + sh[col + 1], 0.0f);
    h.z = (_Float16)fmaxf(z.z * sc[col + 2] + sh[col + 2], 0.0f);
    h.w = (_Float16)fmaxf(z.w * sc[col + 3] + sh[col + 3], 0.0f);
    *(v4h*)(Aout + i4) = h;
}

// h2 = relu(h1 + bn3(z3)) -> f16  (h1 stored as f16 a1, no relu on bn3 term)
__global__ void h2_kernel(const _Float16* __restrict__ A1, const float* __restrict__ Z3,
                          const float* __restrict__ sc, const float* __restrict__ sh,
                          _Float16* __restrict__ Aout, size_t n4) {
    size_t t = (size_t)blockIdx.x * blockDim.x + threadIdx.x;
    if (t >= n4) return;
    size_t i4 = t * 4;
    int col = (int)(i4 & 255);
    v4f z = *(const v4f*)(Z3 + i4);
    v4h a = *(const v4h*)(A1 + i4);
    v4h h;
    h.x = (_Float16)fmaxf((float)a.x + z.x * sc[col + 0] + sh[col + 0], 0.0f);
    h.y = (_Float16)fmaxf((float)a.y + z.y * sc[col + 1] + sh[col + 1], 0.0f);
    h.z = (_Float16)fmaxf((float)a.z + z.z * sc[col + 2] + sh[col + 2], 0.0f);
    h.w = (_Float16)fmaxf((float)a.w + z.w * sc[col + 3] + sh[col + 3], 0.0f);
    *(v4h*)(Aout + i4) = h;
}

// ---------------------------------------------------------------------------
// Polar projection of 3x3 matrices: Newton iteration for the orthogonal polar
// factor UV^T (== SVD's U@Vh), with |det|^(-1/3) scaling for fast convergence,
// then multiply by sign(det) as in the reference.
// ---------------------------------------------------------------------------
__global__ void polar_kernel(const float* __restrict__ outlin, float* __restrict__ rot) {
    int gid = blockIdx.x * blockDim.x + threadIdx.x;
    if (gid >= NROWS * 24) return;
    int row = gid / 24;
    int j   = gid - row * 24;
    const float* p = outlin + (size_t)row * OPAD + j * 9;
    float a00 = p[0], a01 = p[1], a02 = p[2];
    float a10 = p[3], a11 = p[4], a12 = p[5];
    float a20 = p[6], a21 = p[7], a22 = p[8];

    #pragma unroll
    for (int it = 0; it < 12; ++it) {
        float c00 = a11*a22 - a12*a21;
        float c01 = a12*a20 - a10*a22;
        float c02 = a10*a21 - a11*a20;
        float c10 = a21*a02 - a22*a01;
        float c11 = a22*a00 - a20*a02;
        float c12 = a20*a01 - a21*a00;
        float c20 = a01*a12 - a02*a11;
        float c21 = a02*a10 - a00*a12;
        float c22 = a00*a11 - a01*a10;
        float det = a00*c00 + a01*c01 + a02*c02;
        float mu = 1.0f;
        if (it < 6) mu = 1.0f / cbrtf(fmaxf(fabsf(det), 1e-30f));
        float hm = 0.5f * mu;
        float hd = 0.5f / (det * mu);
        a00 = hm*a00 + hd*c00;  a01 = hm*a01 + hd*c01;  a02 = hm*a02 + hd*c02;
        a10 = hm*a10 + hd*c10;  a11 = hm*a11 + hd*c11;  a12 = hm*a12 + hd*c12;
        a20 = hm*a20 + hd*c20;  a21 = hm*a21 + hd*c21;  a22 = hm*a22 + hd*c22;
    }
    float det = a00*(a11*a22 - a12*a21) - a01*(a10*a22 - a12*a20) + a02*(a10*a21 - a11*a20);
    float s = (det < 0.0f) ? -1.0f : 1.0f;
    float* o = rot + (size_t)gid * 9;
    o[0] = a00*s; o[1] = a01*s; o[2] = a02*s;
    o[3] = a10*s; o[4] = a11*s; o[5] = a12*s;
    o[6] = a20*s; o[7] = a21*s; o[8] = a22*s;
}

// betas (10) + camera (3) copy-out
__global__ void tail_kernel(const float* __restrict__ outlin, float* __restrict__ out,
                            size_t betasOff, size_t camOff) {
    int gid = blockIdx.x * blockDim.x + threadIdx.x;
    if (gid >= NROWS * 13) return;
    int row = gid / 13;
    int c   = gid - row * 13;
    float v = outlin[(size_t)row * OPAD + 216 + c];
    if (c < 10) out[betasOff + (size_t)row * 10 + c] = v;
    else        out[camOff   + (size_t)row * 3 + (c - 10)] = v;
}

// ---------------------------------------------------------------------------
extern "C" void kernel_launch(void* const* d_in, const int* in_sizes, int n_in,
                              void* d_out, int out_size, void* d_ws, size_t ws_size,
                              hipStream_t stream) {
    const float* x   = (const float*)d_in[0];
    // d_in[1] = pred_class (unused by reference)
    const float* W1  = (const float*)d_in[2];
    const float* b1  = (const float*)d_in[3];
    const float* g1  = (const float*)d_in[4];
    const float* be1 = (const float*)d_in[5];
    const float* W2  = (const float*)d_in[6];
    const float* b2  = (const float*)d_in[7];
    const float* g2  = (const float*)d_in[8];
    const float* be2 = (const float*)d_in[9];
    const float* W3  = (const float*)d_in[10];
    const float* b3  = (const float*)d_in[11];
    const float* g3  = (const float*)d_in[12];
    const float* be3 = (const float*)d_in[13];
    const float* W4  = (const float*)d_in[14];
    const float* b4  = (const float*)d_in[15];
    float* out = (float*)d_out;

    // ---- workspace layout (bytes, all 256B aligned) ----
    char* ws = (char*)d_ws;
    size_t o = 0;
    _Float16* Wt1 = (_Float16*)(ws + o); o += (size_t)CDIM * CDIM * 2;
    _Float16* Wt2 = (_Float16*)(ws + o); o += (size_t)CDIM * CDIM * 2;
    _Float16* Wt3 = (_Float16*)(ws + o); o += (size_t)CDIM * CDIM * 2;
    _Float16* Wt4 = (_Float16*)(ws + o); o += (size_t)OPAD * CDIM * 2;
    float* stats  = (float*)(ws + o);    o += 6 * CDIM * 4;   // sum1,sq1,sum2,sq2,sum3,sq3
    float* scsh   = (float*)(ws + o);    o += 6 * CDIM * 4;   // sc1,sh1,sc2,sh2,sc3,sh3
    _Float16* aA  = (_Float16*)(ws + o); o += (size_t)NROWS * CDIM * 2;  // a0 -> a2 -> a3
    _Float16* aB  = (_Float16*)(ws + o); o += (size_t)NROWS * CDIM * 2;  // a1
    float* zbuf   = (float*)(ws + o);    o += (size_t)NROWS * CDIM * 4;  // z1/z2/z3 (reused)
    float* outlin = (float*)(ws + o);    o += (size_t)NROWS * OPAD * 4;
    (void)ws_size; (void)n_in; (void)in_sizes; (void)out_size;

    float* sum1 = stats;            float* sq1 = stats + CDIM;
    float* sum2 = stats + 2*CDIM;   float* sq2 = stats + 3*CDIM;
    float* sum3 = stats + 4*CDIM;   float* sq3 = stats + 5*CDIM;
    float* sc1 = scsh;              float* sh1 = scsh + CDIM;
    float* sc2 = scsh + 2*CDIM;     float* sh2 = scsh + 3*CDIM;
    float* sc3 = scsh + 4*CDIM;     float* sh3 = scsh + 5*CDIM;

    const size_t nElem = (size_t)NROWS * CDIM;       // 29,491,200
    const size_t n4    = nElem / 4;                  // 7,372,800
    const int    cvtBlocks  = (int)((n4 + 255) / 256);
    const int    gemmBlocks = NROWS / 64;            // 1800
    const float  invN = 1.0f / (float)NROWS;

    // 1) weight transpose + f16 convert (weights then persist in L2)
    wprep_kernel<<<(CDIM*CDIM + 255)/256, 256, 0, stream>>>(W1, Wt1, CDIM, CDIM);
    wprep_kernel<<<(CDIM*CDIM + 255)/256, 256, 0, stream>>>(W2, Wt2, CDIM, CDIM);
    wprep_kernel<<<(CDIM*CDIM + 255)/256, 256, 0, stream>>>(W3, Wt3, CDIM, CDIM);
    wprep_kernel<<<(OPAD*CDIM + 255)/256, 256, 0, stream>>>(W4, Wt4, ODIM, OPAD);

    // 2) x -> f16, zero BN stats (graph-replay safe: re-zeroed every call)
    cvt_kernel<<<cvtBlocks, 256, 0, stream>>>(x, aA, n4);
    zero_kernel<<<6, 256, 0, stream>>>(stats, 6 * CDIM);

    // 3) layer 1: z1 = a0 @ W1 + b1 (+stats) ; a1 = relu(bn1(z1))
    gemm_f16_kernel<16, true><<<gemmBlocks, 128, 0, stream>>>(aA, Wt1, b1, zbuf, sum1, sq1, CDIM, CDIM);
    bnfin_kernel<<<1, 256, 0, stream>>>(sum1, sq1, g1, be1, sc1, sh1, invN);
    bnrelu_kernel<<<cvtBlocks, 256, 0, stream>>>(zbuf, sc1, sh1, aB, n4);

    // 4) layer 2: z2 = a1 @ W2 + b2 ; a2 = relu(bn2(z2))
    gemm_f16_kernel<16, true><<<gemmBlocks, 128, 0, stream>>>(aB, Wt2, b2, zbuf, sum2, sq2, CDIM, CDIM);
    bnfin_kernel<<<1, 256, 0, stream>>>(sum2, sq2, g2, be2, sc2, sh2, invN);
    bnrelu_kernel<<<cvtBlocks, 256, 0, stream>>>(zbuf, sc2, sh2, aA, n4);

    // 5) layer 3: z3 = a2 @ W3 + b3 ; h2 = relu(a1 + bn3(z3))
    gemm_f16_kernel<16, true><<<gemmBlocks, 128, 0, stream>>>(aA, Wt3, b3, zbuf, sum3, sq3, CDIM, CDIM);
    bnfin_kernel<<<1, 256, 0, stream>>>(sum3, sq3, g3, be3, sc3, sh3, invN);
    h2_kernel<<<cvtBlocks, 256, 0, stream>>>(aB, zbuf, sc3, sh3, aA, n4);

    // 6) layer 4: out = h2 @ W4 + b4 (no BN/stats), padded to 240 cols
    gemm_f16_kernel<15, false><<<gemmBlocks, 128, 0, stream>>>(aA, Wt4, b4, outlin, nullptr, nullptr, OPAD, ODIM);

    // 7) SO(3) projection of 2,764,800 3x3 matrices + betas/camera copy-out
    const size_t betasOff = (size_t)NROWS * 216;          // rotmat elements
    const size_t camOff   = betasOff + (size_t)NROWS * 10;
    polar_kernel<<<(NROWS*24 + 255)/256, 256, 0, stream>>>(outlin, out);
    tail_kernel<<<(NROWS*13 + 255)/256, 256, 0, stream>>>(outlin, out, betasOff, camOff);
}
